// Decoder_10041633538889
// MI455X (gfx1250) — matrix operand order repaired and checked by reading, not compiled, not run on hardware
//
#include <hip/hip_runtime.h>
#include <hip/hip_bf16.h>
#include <math.h>

// ---------------- types for CDNA5 WMMA (wave32) ----------------
typedef __attribute__((ext_vector_type(16))) __bf16 v16bf;
typedef __attribute__((ext_vector_type(8)))  float  v8f;
typedef __attribute__((ext_vector_type(8)))  __bf16 bf16x8;
typedef __attribute__((ext_vector_type(8)))  float  f32x8;

union FragBF { v16bf v; bf16x8 h[2]; };

// ---------------- problem constants ----------------
constexpr int Bn = 128;   // batch
constexpr int Tn = 128;   // time steps
constexpr int Dn = 1024;  // hidden
constexpr int En = 40;    // emb dim
constexpr int Sn = 128;   // slots
constexpr int In = 32;    // intents
constexpr int NG = 4096;            // 4*D gate outputs
constexpr int KIH = En + 2 * Dn;    // 2088 (W_ih inner)
constexpr int KXU = KIH + Dn;       // 3112 (unpadded concat K)
constexpr int KX  = 3136;           // padded to multiple of 32 (K/32 = 98, even)
constexpr int KS  = 2 * Dn;         // 2048 slot/intent inner
constexpr float NEGC = -1e12f;

constexpr int LDS_ROW = 80;         // padded LDS row stride in bytes (32 bf16 -> 64B + 16B pad)

__device__ __forceinline__ float sigf(float x) { return 1.f / (1.f + __expf(-x)); }
__device__ __forceinline__ bf16x8 cvt_bf8(f32x8 v) { return __builtin_convertvector(v, bf16x8); }

// ---------------- generic bf16 WMMA GEMM: C[M,N] = A[M,K] * Bw[N,K]^T (+bias) ----------------
// Each wave computes a 32x32 macrotile as 2x2 WMMA 16x16 tiles, K-chained in steps of 32.
// Used for the one-shot energies GEMM (A in f32, converted in-flight) and the small slot GEMM.
template <bool A_F32, bool OUT_BF16>
__global__ __launch_bounds__(128) void wmma_gemm(
    const void* __restrict__ Ap, const __bf16* __restrict__ Bw,
    const float* __restrict__ bias, void* __restrict__ Cp,
    int M, int N, int K, int lda, int ldb, int ldc) {
  const int wid = blockIdx.x * (blockDim.x >> 5) + (threadIdx.x >> 5);
  const int tn = N >> 5;
  if (wid >= (M >> 5) * tn) return;   // wave-uniform: EXEC stays all-ones for WMMA
  const int mt = wid / tn, nt = wid % tn;
  const int lane = threadIdx.x & 31;
  const int half = lane >> 4, lr = lane & 15;

  const size_t rowA0 = (size_t)(mt * 32 + lr) * lda;
  const size_t rowA1 = rowA0 + (size_t)16 * lda;
  const size_t colB0 = (size_t)(nt * 32 + lr) * ldb;
  const size_t colB1 = colB0 + (size_t)16 * ldb;

  v8f acc00 = {}, acc01 = {}, acc10 = {}, acc11 = {};

  for (int k0 = 0; k0 < K; k0 += 32) {
    // A 16x32 bf16 fragment (ISA 7.12.2): lane<16 rows hold K {0..7,16..23};
    // lanes>=16 hold K {8..15,24..31}  -> two 8-elem chunks at ka and ka+16.
    const int ka = k0 + half * 8;
    // B 32x16 fragment: same K order as A: lane<16 holds K {0..7,16..23}, lane>=16 holds K {8..15,24..31}.
    const int kb = k0 + half * 8;
    FragBF a0, a1, b0, b1;
    if constexpr (A_F32) {
      const float* A = (const float*)Ap;
      a0.h[0] = cvt_bf8(*(const f32x8*)(A + rowA0 + ka));
      a0.h[1] = cvt_bf8(*(const f32x8*)(A + rowA0 + ka + 16));
      a1.h[0] = cvt_bf8(*(const f32x8*)(A + rowA1 + ka));
      a1.h[1] = cvt_bf8(*(const f32x8*)(A + rowA1 + ka + 16));
    } else {
      const __bf16* A = (const __bf16*)Ap;
      a0.h[0] = *(const bf16x8*)(A + rowA0 + ka);
      a0.h[1] = *(const bf16x8*)(A + rowA0 + ka + 16);
      a1.h[0] = *(const bf16x8*)(A + rowA1 + ka);
      a1.h[1] = *(const bf16x8*)(A + rowA1 + ka + 16);
    }
    b0.h[0] = *(const bf16x8*)(Bw + colB0 + kb);
    b0.h[1] = *(const bf16x8*)(Bw + colB0 + kb + 16);
    b1.h[0] = *(const bf16x8*)(Bw + colB1 + kb);
    b1.h[1] = *(const bf16x8*)(Bw + colB1 + kb + 16);

    acc00 = __builtin_amdgcn_wmma_f32_16x16x32_bf16(false, a0.v, false, b0.v, (short)0, acc00, false, false);
    acc01 = __builtin_amdgcn_wmma_f32_16x16x32_bf16(false, a0.v, false, b1.v, (short)0, acc01, false, false);
    acc10 = __builtin_amdgcn_wmma_f32_16x16x32_bf16(false, a1.v, false, b0.v, (short)0, acc10, false, false);
    acc11 = __builtin_amdgcn_wmma_f32_16x16x32_bf16(false, a1.v, false, b1.v, (short)0, acc11, false, false);
  }

  const int c0 = nt * 32 + lr, c1 = c0 + 16;
  const float bb0 = bias ? bias[c0] : 0.f;
  const float bb1 = bias ? bias[c1] : 0.f;
#pragma unroll
  for (int r = 0; r < 8; ++r) {
    const int r0 = mt * 32 + half * 8 + r;  // C/D: VGPR r -> row r / r+8, col = lane&15
    const int r1 = r0 + 16;
    float v00 = acc00[r] + bb0, v01 = acc01[r] + bb1;
    float v10 = acc10[r] + bb0, v11 = acc11[r] + bb1;
    if constexpr (OUT_BF16) {
      __bf16* C = (__bf16*)Cp;
      C[(size_t)r0 * ldc + c0] = (__bf16)v00;
      C[(size_t)r0 * ldc + c1] = (__bf16)v01;
      C[(size_t)r1 * ldc + c0] = (__bf16)v10;
      C[(size_t)r1 * ldc + c1] = (__bf16)v11;
    } else {
      float* C = (float*)Cp;
      C[(size_t)r0 * ldc + c0] = v00;
      C[(size_t)r0 * ldc + c1] = v01;
      C[(size_t)r1 * ldc + c0] = v10;
      C[(size_t)r1 * ldc + c1] = v11;
    }
  }
}

// ---------------- async-staged WMMA GEMM for the hot gates GEMM ----------------
// Block = 4 waves, covers a 32(M) x 256(N) macrotile; each wave does 32x64 (2x4 WMMA tiles).
// The shared 32-row A stripe is double-buffered through LDS with
// GLOBAL_LOAD_ASYNC_TO_LDS_B128 (ASYNCcnt) and read back via ds_load_b128 in WMMA
// fragment layout; LDS rows padded to 80B so the 16 lanes hit disjoint bank groups.
// K loop is 2x-unrolled with ping-pong B fragment register sets (no copies -> no
// WMMA->VALU WAR hazard NOPs).  Requires: M%32==0, N%256==0, K%64==0.
__global__ __launch_bounds__(128) void wmma_gemm_async(
    const __bf16* __restrict__ A, const __bf16* __restrict__ Bw,
    const float* __restrict__ bias, float* __restrict__ C,
    int M, int N, int K, int lda, int ldb, int ldc) {
  __shared__ __align__(16) unsigned char atile[2 * 32 * LDS_ROW];
  const int nb = N >> 8;                   // N / 256
  const int mb = blockIdx.x / nb;
  const int nbb = blockIdx.x % nb;
  const int wave = threadIdx.x >> 5;
  const int lane = threadIdx.x & 31;
  const int half = lane >> 4, lr = lane & 15;
  const int mbase = mb * 32;
  const int nbase = nbb * 256 + wave * 64;

  // staging role: each of the 128 threads async-copies one 16B chunk per K-step
  const int sr = threadIdx.x >> 2;         // row 0..31
  const int sq = threadIdx.x & 3;          // 16B quarter within the 64B row
  const uint32_t lds_base = (uint32_t)(uintptr_t)(&atile[0]);
  const uint32_t stage_dst0 = lds_base + (uint32_t)(sr * LDS_ROW + sq * 16);
  const __bf16* stage_src0 = A + (size_t)(mbase + sr) * lda + sq * 8;

  v8f acc[2][4];
#pragma unroll
  for (int i = 0; i < 2; ++i)
#pragma unroll
    for (int j = 0; j < 4; ++j) acc[i][j] = v8f{};

  size_t colB[4];
#pragma unroll
  for (int j = 0; j < 4; ++j) colB[j] = (size_t)(nbase + j * 16 + lr) * ldb;

  FragBF B0[4], B1[4];

  // one K-step: consume LDS buffer `buf` + B fragments `Bc`, while async-staging
  // A(kcur+32) into buf^1 and loading B(kcur+32) into `Bp`.
  auto kstep = [&](int buf, int kcur, FragBF (&Bc)[4], FragBF (&Bp)[4]) {
    const int nk = kcur + 32;
    if (nk < K) {
      const uint32_t dst = stage_dst0 + (uint32_t)((buf ^ 1) * 32 * LDS_ROW);
      const __bf16* src = stage_src0 + nk;
      asm volatile("global_load_async_to_lds_b128 %0, %1, off"
                   :: "v"(dst), "v"(src) : "memory");
      const int kb = nk + half * 8;
#pragma unroll
      for (int j = 0; j < 4; ++j) {
        Bp[j].h[0] = *(const bf16x8*)(Bw + colB[j] + kb);
        Bp[j].h[1] = *(const bf16x8*)(Bw + colB[j] + kb + 16);
      }
    }
    const unsigned char* tb = atile + buf * (32 * LDS_ROW);
    FragBF a0, a1;
    a0.h[0] = *(const bf16x8*)(tb + lr * LDS_ROW + half * 16);
    a0.h[1] = *(const bf16x8*)(tb + lr * LDS_ROW + half * 16 + 32);
    a1.h[0] = *(const bf16x8*)(tb + (lr + 16) * LDS_ROW + half * 16);
    a1.h[1] = *(const bf16x8*)(tb + (lr + 16) * LDS_ROW + half * 16 + 32);
#pragma unroll
    for (int j = 0; j < 4; ++j) {
      acc[0][j] = __builtin_amdgcn_wmma_f32_16x16x32_bf16(false, a0.v, false, Bc[j].v, (short)0, acc[0][j], false, false);
      acc[1][j] = __builtin_amdgcn_wmma_f32_16x16x32_bf16(false, a1.v, false, Bc[j].v, (short)0, acc[1][j], false, false);
    }
    // drain this wave's async copy, then block barrier: next step reads buf^1 and
    // the step after may overwrite buf.
    asm volatile("s_wait_asynccnt 0x0" ::: "memory");
    __syncthreads();
  };

  {  // prologue: B(k=0) into B0, A(k=0) into LDS buf 0
    const int kb = half * 8;
#pragma unroll
    for (int j = 0; j < 4; ++j) {
      B0[j].h[0] = *(const bf16x8*)(Bw + colB[j] + kb);
      B0[j].h[1] = *(const bf16x8*)(Bw + colB[j] + kb + 16);
    }
    asm volatile("global_load_async_to_lds_b128 %0, %1, off"
                 :: "v"(stage_dst0), "v"(stage_src0) : "memory");
    asm volatile("s_wait_asynccnt 0x0" ::: "memory");
    __syncthreads();
  }

  for (int k0 = 0; k0 < K; k0 += 64) {   // K/32 even: no tail
    kstep(0, k0, B0, B1);
    kstep(1, k0 + 32, B1, B0);
  }

#pragma unroll
  for (int j = 0; j < 4; ++j) {
    const int c = nbase + j * 16 + lr;
    const float bb = bias ? bias[c] : 0.f;
#pragma unroll
    for (int i = 0; i < 2; ++i) {
      const int rb = mbase + i * 16 + half * 8;
#pragma unroll
      for (int r = 0; r < 8; ++r)
        C[(size_t)(rb + r) * ldc + c] = acc[i][j][r] + bb;
    }
  }
}

// ---------------- weight prep ----------------
__global__ void build_wgates(const float* __restrict__ Wih, const float* __restrict__ Whh,
                             __bf16* __restrict__ Wg) {
  size_t idx = (size_t)blockIdx.x * blockDim.x + threadIdx.x;
  if (idx >= (size_t)NG * KX) return;
  int n = (int)(idx / KX), k = (int)(idx % KX);
  float v = (k < KIH) ? Wih[(size_t)n * KIH + k]
            : (k < KXU) ? Whh[(size_t)n * Dn + (k - KIH)] : 0.f;
  Wg[idx] = (__bf16)v;
}

__global__ void f32_to_bf16(const float* __restrict__ src, __bf16* __restrict__ dst, int n) {
  int i = blockIdx.x * blockDim.x + threadIdx.x;
  if (i < n) dst[i] = (__bf16)src[i];
}

__global__ void combine_bias(const float* a, const float* b, float* o, int n) {
  int i = blockIdx.x * blockDim.x + threadIdx.x;
  if (i < n) o[i] = a[i] + b[i];
}

// ---------------- state init ----------------
__global__ void init_state(const int* __restrict__ ids, const float* __restrict__ context,
                           const float* __restrict__ enc, const float* __restrict__ emb,
                           __bf16* __restrict__ xcat, __bf16* __restrict__ ctxb,
                           float* __restrict__ c) {
  int b = blockIdx.x;
  int id = ids[b];
  for (int k = threadIdx.x; k < KX; k += blockDim.x) {
    float v;
    if (k < En)            v = emb[(size_t)id * En + k];
    else if (k < En + Dn)  v = context[(size_t)b * Dn + (k - En)];
    else if (k < KIH)      v = enc[(size_t)b * Tn * Dn + (k - (En + Dn))];  // aligned = enc[:,0,:]
    else                   v = 0.f;                                         // h0 = 0, K pad = 0
    xcat[(size_t)b * KX + k] = (__bf16)v;
  }
  for (int d = threadIdx.x; d < Dn; d += blockDim.x) {
    ctxb[(size_t)b * Dn + d] = (__bf16)context[(size_t)b * Dn + d];
    c[(size_t)b * Dn + d] = 0.f;
  }
}

// ---------------- LSTM elementwise ----------------
__global__ void lstm_elem(const float* __restrict__ gates, float* __restrict__ c,
                          float* __restrict__ h, __bf16* __restrict__ xcat,
                          __bf16* __restrict__ slotcat, const __bf16* __restrict__ ctxb,
                          float* __restrict__ h1, int save_h1) {
  int idx = blockIdx.x * blockDim.x + threadIdx.x;  // Bn*Dn
  int b = idx >> 10, d = idx & (Dn - 1);
  const float* g = gates + (size_t)b * NG;
  float ig = g[d], fg = g[Dn + d], gg = g[2 * Dn + d], og = g[3 * Dn + d];
  float cn = sigf(fg) * c[idx] + sigf(ig) * tanhf(gg);
  float hn = sigf(og) * tanhf(cn);
  c[idx] = cn;
  h[idx] = hn;
  xcat[(size_t)b * KX + KIH + d] = (__bf16)hn;      // h slot for next-step gates
  slotcat[(size_t)b * KS + d] = (__bf16)hn;          // [h, ctx_old] for slot scores
  slotcat[(size_t)b * KS + Dn + d] = ctxb[idx];      // ctx from step entry (pre-update)
  if (save_h1) h1[idx] = hn;
}

// ---------------- slot post: log_softmax + out write + argmax + emb gather ----------------
__global__ void slot_post(const float* __restrict__ score, const float* __restrict__ emb,
                          __bf16* __restrict__ xcat, float* __restrict__ out, int t) {
  int b = blockIdx.x, s = threadIdx.x;  // 128 threads
  __shared__ float red[Sn];
  __shared__ int ridx[Sn];
  __shared__ int winner;
  float v = score[(size_t)b * Sn + s];
  red[s] = v; __syncthreads();
  for (int o = 64; o; o >>= 1) { if (s < o) red[s] = fmaxf(red[s], red[s + o]); __syncthreads(); }
  float mx = red[0]; __syncthreads();
  float e = __expf(v - mx);
  red[s] = e; __syncthreads();
  for (int o = 64; o; o >>= 1) { if (s < o) red[s] += red[s + o]; __syncthreads(); }
  float lse = mx + __logf(red[0]);
  out[((size_t)b * Tn + t) * Sn + s] = v - lse;
  __syncthreads();
  red[s] = v; ridx[s] = s; __syncthreads();
  for (int o = 64; o; o >>= 1) {
    if (s < o) {
      bool take = (red[s + o] > red[s]) || (red[s + o] == red[s] && ridx[s + o] < ridx[s]);
      if (take) { red[s] = red[s + o]; ridx[s] = ridx[s + o]; }
    }
    __syncthreads();
  }
  if (s == 0) winner = ridx[0];
  __syncthreads();
  if (s < En) xcat[(size_t)b * KX + s] = (__bf16)emb[(size_t)winner * En + s];  // emb for next step
}

// ---------------- attention ----------------
// sc[b,t] = dot(energies[b,t,:], h[b,:])   one wave per (b,t)
__global__ void attend_scores(const __bf16* __restrict__ energ, const float* __restrict__ h,
                              const unsigned char* __restrict__ mask, float* __restrict__ sc) {
  int gw = blockIdx.x * (blockDim.x >> 5) + (threadIdx.x >> 5);  // b*Tn + t
  int b = gw >> 7;
  int lane = threadIdx.x & 31;
  const bf16x8* e8 = (const bf16x8*)(energ + (size_t)gw * Dn);
  const f32x8* h8 = (const f32x8*)(h + (size_t)b * Dn);
  float s = 0.f;
  for (int i = lane; i < Dn / 8; i += 32) {
    bf16x8 ev = e8[i];
    f32x8 hv = h8[i];
#pragma unroll
    for (int j = 0; j < 8; ++j) s += (float)ev[j] * hv[j];
  }
  for (int o = 16; o; o >>= 1) s += __shfl_down(s, o, 32);
  if (lane == 0) sc[gw] = mask[gw] ? NEGC : s;
}

__global__ void attend_softmax(const float* __restrict__ sc, float* __restrict__ a) {
  int b = blockIdx.x, t = threadIdx.x;  // 128 threads
  __shared__ float red[Tn];
  float v = sc[(size_t)b * Tn + t];
  red[t] = v; __syncthreads();
  for (int o = 64; o; o >>= 1) { if (t < o) red[t] = fmaxf(red[t], red[t + o]); __syncthreads(); }
  float mx = red[0]; __syncthreads();
  float e = __expf(v - mx);
  red[t] = e; __syncthreads();
  for (int o = 64; o; o >>= 1) { if (t < o) red[t] += red[t + o]; __syncthreads(); }
  a[(size_t)b * Tn + t] = e / red[0];
}

// ctx[b,d] = sum_t a[b,t] * enc[b,t,d]; outputs optional (f32 / xcat ctx slot / bf16 buf)
__global__ void attend_ctx(const float* __restrict__ a, const float* __restrict__ enc,
                           float* __restrict__ outf, __bf16* __restrict__ xcat,
                           __bf16* __restrict__ ctxb) {
  int b = blockIdx.x;
  __shared__ float aw[Tn];
  for (int t = threadIdx.x; t < Tn; t += blockDim.x) aw[t] = a[(size_t)b * Tn + t];
  __syncthreads();
  for (int d = threadIdx.x; d < Dn; d += blockDim.x) {
    const float* eb = enc + (size_t)b * Tn * Dn + d;
    float s = 0.f;
    for (int t = 0; t < Tn; ++t) s += aw[t] * eb[(size_t)t * Dn];
    if (outf) outf[(size_t)b * Dn + d] = s;
    if (xcat) xcat[(size_t)b * KX + En + d] = (__bf16)s;
    if (ctxb) ctxb[(size_t)b * Dn + d] = (__bf16)s;
  }
}

__global__ void set_aligned(const float* __restrict__ enc, __bf16* __restrict__ xcat, int t) {
  int b = blockIdx.x;
  for (int d = threadIdx.x; d < Dn; d += blockDim.x)
    xcat[(size_t)b * KX + En + Dn + d] = (__bf16)enc[((size_t)b * Tn + t) * Dn + d];
}

// ---------------- intent head (tiny, fp32) ----------------
__global__ void intent_k(const float* __restrict__ h1, const float* __restrict__ ictx,
                         const float* __restrict__ Wi, const float* __restrict__ bi,
                         float* __restrict__ out) {
  int b = blockIdx.x, i = threadIdx.x;  // 32 threads
  float acc = bi[i];
  const float* w = Wi + (size_t)i * KS;
  const float* hb = h1 + (size_t)b * Dn;
  const float* cb = ictx + (size_t)b * Dn;
  for (int k = 0; k < Dn; ++k) acc += hb[k] * w[k];
  for (int k = 0; k < Dn; ++k) acc += cb[k] * w[Dn + k];
  out[(size_t)Bn * Tn * Sn + (size_t)b * In + i] = acc;
}

// ---------------- host ----------------
static inline size_t alup(size_t x) { return (x + 4095) & ~(size_t)4095; }

extern "C" void kernel_launch(void* const* d_in, const int* in_sizes, int n_in,
                              void* d_out, int out_size, void* d_ws, size_t ws_size,
                              hipStream_t stream) {
  const int* ids = (const int*)d_in[0];
  const float* context = (const float*)d_in[1];
  const float* enc = (const float*)d_in[2];
  const unsigned char* mask = (const unsigned char*)d_in[3];
  const float* emb = (const float*)d_in[4];
  const float* W_ih = (const float*)d_in[5];
  const float* b_ih = (const float*)d_in[6];
  const float* W_hh = (const float*)d_in[7];
  const float* b_hh = (const float*)d_in[8];
  const float* W_attn = (const float*)d_in[9];
  const float* b_attn = (const float*)d_in[10];
  const float* W_slot = (const float*)d_in[11];
  const float* b_slot = (const float*)d_in[12];
  const float* W_intent = (const float*)d_in[13];
  const float* b_intent = (const float*)d_in[14];
  float* out = (float*)d_out;

  // workspace carve-up (~68 MB, all 4 KB aligned)
  char* p = (char*)d_ws;
  size_t off = 0;
  auto take = [&](size_t bytes) { char* r = p + off; off += alup(bytes); return (void*)r; };
  __bf16* Wg     = (__bf16*)take((size_t)NG * KX * 2);         // concat gate weights bf16
  __bf16* WslotB = (__bf16*)take((size_t)Sn * KS * 2);
  __bf16* WattnB = (__bf16*)take((size_t)Dn * Dn * 2);
  float*  biasG  = (float*) take((size_t)NG * 4);
  __bf16* energ  = (__bf16*)take((size_t)Bn * Tn * Dn * 2);    // energies, bf16
  __bf16* xcat   = (__bf16*)take((size_t)Bn * KX * 2);         // [emb|ctx|aligned|h|pad]
  __bf16* slotc  = (__bf16*)take((size_t)Bn * KS * 2);         // [h|ctx_old]
  __bf16* ctxb   = (__bf16*)take((size_t)Bn * Dn * 2);
  float*  gates  = (float*) take((size_t)Bn * NG * 4);
  float*  hbuf   = (float*) take((size_t)Bn * Dn * 4);
  float*  cbuf   = (float*) take((size_t)Bn * Dn * 4);
  float*  h1buf  = (float*) take((size_t)Bn * Dn * 4);
  float*  ictx   = (float*) take((size_t)Bn * Dn * 4);
  float*  score  = (float*) take((size_t)Bn * Sn * 4);
  float*  scb    = (float*) take((size_t)Bn * Tn * 4);
  float*  awb    = (float*) take((size_t)Bn * Tn * 4);
  (void)ws_size; (void)in_sizes; (void)n_in; (void)out_size;

  // --- weight prep ---
  {
    size_t n = (size_t)NG * KX;
    build_wgates<<<(unsigned)((n + 255) / 256), 256, 0, stream>>>(W_ih, W_hh, Wg);
    f32_to_bf16<<<(Sn * KS + 255) / 256, 256, 0, stream>>>(W_slot, WslotB, Sn * KS);
    f32_to_bf16<<<(Dn * Dn + 255) / 256, 256, 0, stream>>>(W_attn, WattnB, Dn * Dn);
    combine_bias<<<(NG + 255) / 256, 256, 0, stream>>>(b_ih, b_hh, biasG, NG);
  }

  // --- energies = enc @ W_attn^T + b_attn  (f32 A converted in-kernel, bf16 out) ---
  {
    int waves = (Bn * Tn / 32) * (Dn / 32);  // 16384
    wmma_gemm<true, true><<<waves / 4, 128, 0, stream>>>(
        enc, WattnB, b_attn, energ, Bn * Tn, Dn, Dn, Dn, Dn, Dn);
  }

  // --- initial state ---
  init_state<<<Bn, 256, 0, stream>>>(ids, context, enc, emb, xcat, ctxb, cbuf);

  // --- sequential scan over T ---
  for (int t = 0; t < Tn; ++t) {
    {  // gate GEMM: (128 x 4096 x 3136) with async LDS staging of A
      int blocks = (Bn / 32) * (NG / 256);  // 64
      wmma_gemm_async<<<blocks, 128, 0, stream>>>(
          xcat, Wg, biasG, gates, Bn, NG, KX, KX, KX, NG);
    }
    lstm_elem<<<(Bn * Dn) / 256, 256, 0, stream>>>(gates, cbuf, hbuf, xcat, slotc, ctxb,
                                                   h1buf, t == 0 ? 1 : 0);
    {  // slot GEMM: (128 x 128 x 2048)
      int waves = (Bn / 32) * (Sn / 32);  // 16
      wmma_gemm<false, false><<<waves / 4, 128, 0, stream>>>(
          slotc, WslotB, b_slot, score, Bn, Sn, KS, KS, KS, Sn);
    }
    slot_post<<<Bn, Sn, 0, stream>>>(score, emb, xcat, out, t);
    attend_scores<<<(Bn * Tn) / 4, 128, 0, stream>>>(energ, hbuf, mask, scb);
    attend_softmax<<<Bn, Tn, 0, stream>>>(scb, awb);
    attend_ctx<<<Bn, 256, 0, stream>>>(awb, enc, nullptr, xcat, ctxb);
    if (t + 1 < Tn) set_aligned<<<Bn, 256, 0, stream>>>(enc, xcat, t + 1);
  }

  // --- intent head: ictx = attend(h1); intent = [h1|ictx] @ W_intent^T + b ---
  attend_scores<<<(Bn * Tn) / 4, 128, 0, stream>>>(energ, h1buf, mask, scb);
  attend_softmax<<<Bn, Tn, 0, stream>>>(scb, awb);
  attend_ctx<<<Bn, 256, 0, stream>>>(awb, enc, ictx, nullptr, nullptr);
  intent_k<<<Bn, In, 0, stream>>>(h1buf, ictx, W_intent, b_intent, out);
}
